// ScaledDotProductAttention_75161927680629
// MI455X (gfx1250) — matrix operand order repaired and checked by reading, not compiled, not run on hardware
//
#include <hip/hip_runtime.h>
#include <hip/hip_bf16.h>

// ---------------------------------------------------------------------------
// Multi-head attention for MI455X (gfx1250), wave32 + WMMA bf16.
//   D_MODEL=1024, H=16, DK=DV=64, B=2, NQ=NK=2048
// Pipeline:
//   1) cvt f32->bf16 (queries/keys/values, Wq/Wk/Wv/Wo)      [elementwise]
//   2) gemm_nt bf16 WMMA: Q,K (head-major), V (transposed)   [async->LDS B]
//   3) flash attention: QK^T wmma -> *weights, mask, online softmax -> P@V
//   4) gemm_nt bf16 WMMA: out = O @ Wo^T + bo -> f32 d_out
// ---------------------------------------------------------------------------

typedef __bf16 bf16_t;
typedef __attribute__((ext_vector_type(16))) __bf16 v16bf_t;
typedef __attribute__((ext_vector_type(8)))  __bf16 v8bf_t;
typedef __attribute__((ext_vector_type(8)))  float  v8f_t;

// Types for the async-copy builtin: AS1/AS3 pointers to int 4-vector,
// exactly matching the builtin's parameter types from the clang diagnostic.
typedef int v4i_t __attribute__((vector_size(16)));
typedef __attribute__((address_space(1))) v4i_t* gas1_v4i_t;
typedef __attribute__((address_space(3))) v4i_t* las3_v4i_t;

#define D_MODEL 1024
#define NHEAD   16
#define DK      64
#define NQ      2048
#define NK      2048
#define NB      2
#define MROWS   (NB * NQ)   // 4096

static __device__ __forceinline__ v16bf_t cat16(v8bf_t lo, v8bf_t hi) {
  v16bf_t r;
#pragma unroll
  for (int i = 0; i < 8; ++i) { r[i] = lo[i]; r[i + 8] = hi[i]; }
  return r;
}

// A-operand (16x32, MxK): lane m holds K in [8h,8h+8) U [16+8h,16+8h+8)
static __device__ __forceinline__ v16bf_t load_a_op(const bf16_t* p, int half) {
  return cat16(*(const v8bf_t*)(p + 8 * half),
               *(const v8bf_t*)(p + 16 + 8 * half));
}
// B-operand (32x16, KxN): lane n holds K in [8h,8h+8) U [16+8h,16+8h+8)
static __device__ __forceinline__ v16bf_t load_b_op(const bf16_t* p, int half) {
  const bf16_t* q = p + 8 * half;
  return cat16(*(const v8bf_t*)q, *(const v8bf_t*)(q + 16));
}

static __device__ __forceinline__ v8f_t wmma_bf16(v16bf_t a, v16bf_t b, v8f_t c) {
  return __builtin_amdgcn_wmma_f32_16x16x32_bf16(false, a, false, b,
                                                 (short)0, c, false, false);
}

// reductions across the 16 lanes sharing a row-set (xor masks stay in-half)
static __device__ __forceinline__ float redmax16(float v) {
  v = fmaxf(v, __shfl_xor(v, 1, 32));
  v = fmaxf(v, __shfl_xor(v, 2, 32));
  v = fmaxf(v, __shfl_xor(v, 4, 32));
  v = fmaxf(v, __shfl_xor(v, 8, 32));
  return v;
}
static __device__ __forceinline__ float redsum16(float v) {
  v += __shfl_xor(v, 1, 32);
  v += __shfl_xor(v, 2, 32);
  v += __shfl_xor(v, 4, 32);
  v += __shfl_xor(v, 8, 32);
  return v;
}

// ---------------------------------------------------------------------------
__global__ void cvt_f32_bf16(const float* __restrict__ x, bf16_t* __restrict__ y,
                             int n) {
  int i = blockIdx.x * blockDim.x + threadIdx.x;
  int stride = gridDim.x * blockDim.x;
  for (; i < n; i += stride) y[i] = (bf16_t)x[i];
}

// ---------------------------------------------------------------------------
// C[m,n] = sum_k A[m,k] * W[n,k] + bias[n]   (A: MROWSx1024, W: 1024x1024)
// mode 0: bf16 out[((b*16+h)*2048 + q)*64 + dk]      (head-major Q/K)
// mode 1: bf16 out[((b*16+h)*64 + dv)*2048 + krow]   (transposed V)
// mode 2: f32  out[m*1024 + n]                       (final projection)
// block: 256 thr = 8 waves; tile 128m x 64n; K chunk 32.
// B tile staged memory->LDS with the CDNA5 async copy engine (ASYNCcnt).
__global__ __launch_bounds__(256) void gemm_nt(const bf16_t* __restrict__ A,
                                               const bf16_t* __restrict__ W,
                                               const float* __restrict__ bias,
                                               void* __restrict__ outp, int mode) {
  __shared__ bf16_t ldsB[64 * 32];
  const int lane = threadIdx.x & 31;
  const int wave = threadIdx.x >> 5;
  const int n16  = lane & 15;
  const int half = lane >> 4;
  const int ntile = blockIdx.x * 64;
  const int mtile = blockIdx.y * 128;

  const bf16_t* arow = A + (size_t)(mtile + wave * 16 + n16) * D_MODEL;

  v8f_t acc[4];
#pragma unroll
  for (int t = 0; t < 4; ++t)
#pragma unroll
    for (int j = 0; j < 8; ++j) acc[t][j] = 0.0f;

  const int r   = threadIdx.x >> 2;   // 0..63  (n row of B tile)
  const int seg = threadIdx.x & 3;    // 0..3   (16B segment within 32-k row)

  for (int kc = 0; kc < D_MODEL; kc += 32) {
    // stage 64x32 B tile: global -> LDS direct (async DMA, no VGPR bounce)
    {
      const bf16_t* src = W + (size_t)(ntile + r) * D_MODEL + kc + seg * 8;
      __builtin_amdgcn_global_load_async_to_lds_b128(
          (gas1_v4i_t)(void*)src,
          (las3_v4i_t)(void*)&ldsB[r * 32 + seg * 8],
          0, 0);
    }
    asm volatile("s_wait_asynccnt 0x0" ::: "memory");
    __syncthreads();

    v16bf_t a = load_a_op(arow + kc, half);
#pragma unroll
    for (int t = 0; t < 4; ++t) {
      v16bf_t b = load_b_op(&ldsB[(16 * t + n16) * 32], half);
      acc[t] = wmma_bf16(a, b, acc[t]);
    }
    __syncthreads();
  }

#pragma unroll
  for (int t = 0; t < 4; ++t) {
    int n = ntile + 16 * t + n16;
    float bv = bias[n];
#pragma unroll
    for (int i = 0; i < 8; ++i) {
      float v = acc[t][i] + bv;
      int m = mtile + wave * 16 + i + 8 * half;
      if (mode == 0) {
        int b = m >> 11, q = m & 2047, h = n >> 6, dk = n & 63;
        ((bf16_t*)outp)[(((size_t)(b * NHEAD + h) * NQ + q) << 6) + dk] = (bf16_t)v;
      } else if (mode == 1) {
        int b = m >> 11, krow = m & 2047, h = n >> 6, dv = n & 63;
        ((bf16_t*)outp)[((size_t)(b * NHEAD + h) * DK + dv) * NK + krow] = (bf16_t)v;
      } else {
        ((float*)outp)[(size_t)m * D_MODEL + n] = v;
      }
    }
  }
}

// ---------------------------------------------------------------------------
// Flash attention. Block = 256 thr (8 waves); each wave owns 16 q rows.
// grid.x = NB * NHEAD * (NQ/128) = 512.
__global__ __launch_bounds__(256) void mha_attn(const bf16_t* __restrict__ Qb,
                                                const bf16_t* __restrict__ Kb,
                                                const bf16_t* __restrict__ Vt,
                                                const float* __restrict__ AW,
                                                const unsigned char* __restrict__ AM,
                                                bf16_t* __restrict__ Oc) {
  __shared__ bf16_t ldsP[8][16][32];   // per-wave P transpose buffer
  const int lane = threadIdx.x & 31;
  const int wave = threadIdx.x >> 5;
  const int n16  = lane & 15;
  const int half = lane >> 4;

  int bid  = blockIdx.x;
  int qblk = bid & 15;
  int h    = (bid >> 4) & 15;
  int b    = bid >> 8;
  int qbase = qblk * 128 + wave * 16;
  int bh = b * NHEAD + h;

  // Q A-operands for full DK=64 (two K=32 chunks), resident all loop
  const bf16_t* qrow = Qb + ((size_t)bh * NQ + qbase + n16) * DK;
  v16bf_t aq0 = load_a_op(qrow, half);
  v16bf_t aq1 = load_a_op(qrow + 32, half);

  const bf16_t* kbase = Kb + (size_t)bh * NK * DK;
  const bf16_t* vbase = Vt + (size_t)bh * DK * NK;
  size_t awbase = ((size_t)bh * NQ + qbase) * NK;

  float mrun[8], lrun[8];
  v8f_t oacc[4];
#pragma unroll
  for (int i = 0; i < 8; ++i) { mrun[i] = -3.0e38f; lrun[i] = 0.0f; }
#pragma unroll
  for (int t = 0; t < 4; ++t)
#pragma unroll
    for (int j = 0; j < 8; ++j) oacc[t][j] = 0.0f;

  for (int kc = 0; kc < NK; kc += 32) {
    // ---- S = Q K^T for two 16-wide column tiles ----
    v8f_t s[2];
#pragma unroll
    for (int t = 0; t < 2; ++t) {
      const bf16_t* krow = kbase + (size_t)(kc + 16 * t + n16) * DK;
      v16bf_t bk0 = load_b_op(krow, half);
      v16bf_t bk1 = load_b_op(krow + 32, half);
      v8f_t z;
#pragma unroll
      for (int j = 0; j < 8; ++j) z[j] = 0.0f;
      z = wmma_bf16(aq0, bk0, z);
      s[t] = wmma_bf16(aq1, bk1, z);
    }

    // ---- scale * weights, mask ----
    float sv[2][8];
#pragma unroll
    for (int t = 0; t < 2; ++t)
#pragma unroll
      for (int i = 0; i < 8; ++i) {
        size_t wi = awbase + (size_t)(i + 8 * half) * NK + (kc + 16 * t + n16);
        float w = AW[wi];
        unsigned char mk = AM[wi];
        float x = s[t][i] * 0.125f * w;
        sv[t][i] = mk ? -1.0e9f : x;
      }

    // ---- online softmax (row stats across 16 lanes of each half) ----
    float mnew[8], alpha[8], p0[8], p1[8];
#pragma unroll
    for (int i = 0; i < 8; ++i) {
      float v = redmax16(fmaxf(sv[0][i], sv[1][i]));
      mnew[i]  = fmaxf(mrun[i], v);
      alpha[i] = __expf(mrun[i] - mnew[i]);
      mrun[i]  = mnew[i];
      p0[i] = __expf(sv[0][i] - mnew[i]);
      p1[i] = __expf(sv[1][i] - mnew[i]);
      float rs = redsum16(p0[i] + p1[i]);
      lrun[i] = lrun[i] * alpha[i] + rs;
    }
#pragma unroll
    for (int t = 0; t < 4; ++t)
#pragma unroll
      for (int i = 0; i < 8; ++i) oacc[t][i] *= alpha[i];

    // ---- transpose P (C-layout -> A-layout) through private-wave LDS ----
#pragma unroll
    for (int i = 0; i < 8; ++i) {
      int m = i + 8 * half;
      ldsP[wave][m][n16]      = (bf16_t)p0[i];
      ldsP[wave][m][16 + n16] = (bf16_t)p1[i];
    }
    asm volatile("s_wait_dscnt 0x0" ::: "memory");  // wave-internal DS RAW order
    v16bf_t ap = load_a_op(&ldsP[wave][n16][0], half);

    // ---- O += P @ V  (V stored transposed: rows dv, contiguous k) ----
#pragma unroll
    for (int t = 0; t < 4; ++t) {
      const bf16_t* vrow = vbase + (size_t)(16 * t + n16) * NK + kc;
      v16bf_t bv = load_b_op(vrow, half);
      oacc[t] = wmma_bf16(ap, bv, oacc[t]);
    }

    // prefetch next weights chunk (the one true HBM stream)
    if (kc + 32 < NK)
      __builtin_prefetch(AW + awbase + (size_t)n16 * NK + kc + 32, 0, 1);
  }

  // ---- normalize and write O[b, q, h*64+dv] as bf16 ----
  float inv[8];
#pragma unroll
  for (int i = 0; i < 8; ++i) inv[i] = 1.0f / lrun[i];
#pragma unroll
  for (int t = 0; t < 4; ++t)
#pragma unroll
    for (int i = 0; i < 8; ++i) {
      int row = qbase + i + 8 * half;
      size_t oi = ((size_t)(b * NQ + row)) * D_MODEL + h * DK + 16 * t + n16;
      Oc[oi] = (bf16_t)(oacc[t][i] * inv[i]);
    }
}

// ---------------------------------------------------------------------------
extern "C" void kernel_launch(void* const* d_in, const int* in_sizes, int n_in,
                              void* d_out, int out_size, void* d_ws, size_t ws_size,
                              hipStream_t stream) {
  const float* queries = (const float*)d_in[0];
  const float* keys    = (const float*)d_in[1];
  const float* values  = (const float*)d_in[2];
  const float* attw    = (const float*)d_in[3];
  const unsigned char* attm = (const unsigned char*)d_in[4];  // bool mask
  const float* Wq = (const float*)d_in[5];
  const float* bq = (const float*)d_in[6];
  const float* Wk = (const float*)d_in[7];
  const float* bk = (const float*)d_in[8];
  const float* Wv = (const float*)d_in[9];
  const float* bv = (const float*)d_in[10];
  const float* Wo = (const float*)d_in[11];
  const float* bo = (const float*)d_in[12];

  char* ws = (char*)d_ws;
  const size_t MB = 1024 * 1024;
  bf16_t* Qb  = (bf16_t*)(ws + 0 * MB);    // (B,H,NQ,DK)  8MB
  bf16_t* Kbt = (bf16_t*)(ws + 8 * MB);    // (B,H,NK,DK)  8MB
  bf16_t* Vt  = (bf16_t*)(ws + 16 * MB);   // (B,H,DV,NK)  8MB
  bf16_t* Xq  = (bf16_t*)(ws + 24 * MB);   // bf16 queries 8MB
  bf16_t* Xk  = (bf16_t*)(ws + 32 * MB);
  bf16_t* Xv  = (bf16_t*)(ws + 40 * MB);
  bf16_t* Wqb = (bf16_t*)(ws + 48 * MB);   // 2MB each
  bf16_t* Wkb = (bf16_t*)(ws + 50 * MB);
  bf16_t* Wvb = (bf16_t*)(ws + 52 * MB);
  bf16_t* Wob = (bf16_t*)(ws + 54 * MB);
  bf16_t* Oc  = (bf16_t*)(ws + 56 * MB);   // (B,NQ,H*DV)  8MB

  const int nX = MROWS * D_MODEL;      // 4194304
  const int nW = D_MODEL * D_MODEL;    // 1048576
  cvt_f32_bf16<<<4096, 256, 0, stream>>>(queries, Xq, nX);
  cvt_f32_bf16<<<4096, 256, 0, stream>>>(keys,    Xk, nX);
  cvt_f32_bf16<<<4096, 256, 0, stream>>>(values,  Xv, nX);
  cvt_f32_bf16<<<1024, 256, 0, stream>>>(Wq, Wqb, nW);
  cvt_f32_bf16<<<1024, 256, 0, stream>>>(Wk, Wkb, nW);
  cvt_f32_bf16<<<1024, 256, 0, stream>>>(Wv, Wvb, nW);
  cvt_f32_bf16<<<1024, 256, 0, stream>>>(Wo, Wob, nW);

  dim3 ggrid(D_MODEL / 64, MROWS / 128);  // (16, 32)
  gemm_nt<<<ggrid, 256, 0, stream>>>(Xq, Wqb, bq, (void*)Qb,  0);
  gemm_nt<<<ggrid, 256, 0, stream>>>(Xk, Wkb, bk, (void*)Kbt, 0);
  gemm_nt<<<ggrid, 256, 0, stream>>>(Xv, Wvb, bv, (void*)Vt,  1);

  mha_attn<<<NB * NHEAD * (NQ / 128), 256, 0, stream>>>(Qb, Kbt, Vt, attw, attm, Oc);

  gemm_nt<<<ggrid, 256, 0, stream>>>(Oc, Wob, bo, d_out, 2);
}
